// PriceLSTM_63204738728281
// MI455X (gfx1250) — matrix operand order repaired and checked by reading, not compiled, not run on hardware
//
#include <hip/hip_runtime.h>

// LSTM: SEQ=1024, BATCH=8192, IN=3, HID=50.  Persistent fused kernel:
// each block owns 16 batch rows for the whole time loop; recurrent GEMM
// done with v_wmma_f32_16x16x32_bf16 (f32 accumulate); h/c master state f32,
// c lives purely in registers.  Barriers are LDS-only (s_wait_dscnt +
// s_barrier_signal/wait) so global x prefetches stay in flight across steps;
// x is staged 8 timesteps ahead through a thread-private LDS pipeline.

#define T_SEQ 1024
#define B_SZ  8192
#define IN_SZ 3
#define HID   50
#define NP    256      // padded 4H (4 gates x 64)
#define KP    64       // padded K: 50 (h) + 3 (x) + 11 zero
#define MB    16       // batch rows per block
#define GSTR  20       // gatesT row stride in floats (80B: 16B-aligned, bank-spread)
#define XCH   8        // x prefetch chunk (timesteps)
#define B3    (B_SZ * IN_SZ)

typedef __attribute__((ext_vector_type(16))) __bf16       v16bf;
typedef __attribute__((ext_vector_type(8)))  __bf16       v8bf;
typedef __attribute__((ext_vector_type(8)))  float        v8f;
typedef __attribute__((ext_vector_type(4)))  float        v4f;
typedef __attribute__((ext_vector_type(4)))  unsigned int u32x4;

__device__ __forceinline__ __bf16 f2bf(float f) {
    unsigned u = __builtin_bit_cast(unsigned, f);
    u += 0x7FFFu + ((u >> 16) & 1u);             // round-to-nearest-even
    unsigned short h = (unsigned short)(u >> 16);
    return __builtin_bit_cast(__bf16, h);
}
// v_rcp_f32-based activations (no IEEE div sequence)
__device__ __forceinline__ float sigm(float x)   { return __builtin_amdgcn_rcpf(1.0f + __expf(-x)); }
__device__ __forceinline__ float tanh_f(float x) { return 2.0f * sigm(2.0f * x) - 1.0f; }

// Workgroup barrier that orders LDS only: does NOT drain loadcnt, so global
// prefetches issued before it remain in flight (CDNA5 split counters).
__device__ __forceinline__ void sync_lds() {
    asm volatile("s_wait_dscnt 0x0\n\t"
                 "s_barrier_signal -1\n\t"
                 "s_barrier_wait -1" ::: "memory");
}

__global__ __launch_bounds__(256)
void lstm_persist(const float* __restrict__ x,
                  const float* __restrict__ W_ih,
                  const float* __restrict__ W_hh,
                  const float* __restrict__ b_ih,
                  const float* __restrict__ b_hh,
                  const float* __restrict__ fc_w,
                  const float* __restrict__ fc_b,
                  float* __restrict__ out)
{
    __shared__ __bf16 hbuf[MB][KP];        // A operand: h_t (k<50), x_t (k=50..52), 0-pad
    __shared__ float  gatesT[NP * GSTR];   // gate pre-activations, [n][m]
    __shared__ float  hF[MB][HID + 2];     // final-step hidden state, f32
    __shared__ float  xstage[2][XCH][MB * IN_SZ];  // thread-private x pipeline (cols = tid)

    const int tid   = threadIdx.x;
    const int lane  = tid & 31;
    const int wave  = tid >> 5;            // 0..7, owns N in [wave*32, wave*32+32)
    const int l16   = lane & 15;
    const int lh    = lane >> 4;           // lane half (WMMA layout selector)
    const int bBase = blockIdx.x * MB;

    // ---------------- init LDS state ----------------
    for (int i = tid; i < MB * KP; i += 256) ((__bf16*)hbuf)[i] = f2bf(0.0f);

    // x staging: thread tid<48 owns (row xm, component xi) for all t
    const bool  isx = (tid < MB * IN_SZ);
    const int   xm  = tid / IN_SZ;
    const int   xi  = tid - xm * IN_SZ;
    const float* xp = x + ((size_t)(bBase + xm) * IN_SZ + xi);   // t=0 element
    if (isx) {
        hbuf[xm][HID + xi] = f2bf(xp[0]);                        // stage x_0
        for (int e = 0; e < XCH; ++e)                            // chunk 0 = x_{1..8}
            xstage[0][e][tid] = xp[(size_t)(1 + e) * B3];
    }

    // elementwise ownership: thread owns (em, j = ejb + 16q), q = 0..3
    const int em  = tid & 15;
    const int ejb = tid >> 4;              // 0..15
    float creg[4] = {0.0f, 0.0f, 0.0f, 0.0f};

    // ------- time-invariant B / bias fragments (registers for all 1024 steps) -------
    // B layout (16-bit, 32x16): lanes 0-15: N=Nbase+lane, K=kh*32+{0..15};
    //                           lanes 16-31: same N, K=kh*32+{16..31}.
    v16bf Bf[2][2];
    v8f   biasC[2];
    for (int nt = 0; nt < 2; ++nt) {
        const int n    = wave * 32 + nt * 16 + l16;   // global gate-output column
        const int gate = n >> 6;
        const int j    = n & 63;
        const int row  = gate * HID + j;              // row of W_ih/W_hh (valid iff j<50)
        const float bv = (j < HID) ? (b_ih[row] + b_hh[row]) : 0.0f;
        for (int e = 0; e < 8; ++e) biasC[nt][e] = bv;  // C depends only on N -> splat
        for (int kh = 0; kh < 2; ++kh) {
            const int kb = kh * 32 + lh * 16;
            v16bf bf;
            for (int e = 0; e < 16; ++e) {
                const int k = kb + e;
                float v = 0.0f;
                if (j < HID) {
                    if (k < HID)          v = W_hh[row * HID + k];
                    else if (k < HID + 3) v = W_ih[row * IN_SZ + (k - HID)];
                }
                bf[e] = f2bf(v);
            }
            Bf[nt][kh] = bf;
        }
    }

    // ---------------- recurrence over T (chunks of XCH steps) ----------------
    for (int tc = 0; tc < T_SEQ / XCH; ++tc) {
        // Issue global loads for chunk tc+1 (x indices (tc+1)*8+1 .. +8) NOW;
        // they are committed to LDS only at the END of this chunk, so the
        // loadcnt wait lands ~8 steps' worth of work after issue.
        float xl[XCH];
        if (isx) {
            const float* q = xp + (size_t)((tc + 1) * XCH + 1) * B3;
#pragma unroll
            for (int e = 0; e < XCH; ++e) {
                const int idx = (tc + 1) * XCH + 1 + e;
                xl[e] = (idx < T_SEQ) ? q[(size_t)e * B3] : 0.0f;
            }
        }
        const int par = tc & 1;

        for (int e8 = 0; e8 < XCH; ++e8) {
            const int t = tc * XCH + e8;

            sync_lds();   // hbuf now holds [h_t | x_t]

            // A fragments (16-bit 16x32): lane m=l16; elem e -> k = kh*32+lh*16+e
            v16bf A[2];
            {
                const int m = l16;
#pragma unroll
                for (int kh = 0; kh < 2; ++kh) {
                    const int kb = kh * 32 + lh * 16;
                    const u32x4 lo = *(const u32x4*)&hbuf[m][kb];        // k = kb..kb+7
                    const u32x4 hi = *(const u32x4*)&hbuf[m][kb + 8];    // k = kb+8..kb+15
                    const v8bf blo = __builtin_bit_cast(v8bf, lo);
                    const v8bf bhi = __builtin_bit_cast(v8bf, hi);
                    v16bf a;
#pragma unroll
                    for (int e = 0; e < 8; ++e) { a[e] = blo[e]; a[e + 8] = bhi[e]; }
                    A[kh] = a;
                }
            }

            // gates = [h|x|pad] @ [W_hh;W_ih;0]^T + bias : 2 N-tiles x 2 K-halves
#pragma unroll
            for (int nt = 0; nt < 2; ++nt) {
                v8f acc = biasC[nt];
                acc = __builtin_amdgcn_wmma_f32_16x16x32_bf16(false, A[0], false, Bf[nt][0],
                                                              (short)0, acc, false, false);
                acc = __builtin_amdgcn_wmma_f32_16x16x32_bf16(false, A[1], false, Bf[nt][1],
                                                              (short)0, acc, false, false);
                // C/D layout: elem r -> M = lh*8 + r, N = Nbase + l16
                const int n = wave * 32 + nt * 16 + l16;
                float* gp = &gatesT[n * GSTR + lh * 8];
                *(v4f*)gp       = __builtin_shufflevector(acc, acc, 0, 1, 2, 3);
                *(v4f*)(gp + 4) = __builtin_shufflevector(acc, acc, 4, 5, 6, 7);
            }

            sync_lds();   // gate pre-activations ready

            // elementwise LSTM cell update, f32, c in registers (gate order i,f,g,o)
#pragma unroll
            for (int q = 0; q < 4; ++q) {
                const int j = ejb + 16 * q;            // compile-time offsets from one base
                if (q < 3 || ejb < 2) {                // j < 50
                    const float gi = sigm  (gatesT[(j)       * GSTR + em]);
                    const float gf = sigm  (gatesT[(64 + j)  * GSTR + em]);
                    const float gg = tanh_f(gatesT[(128 + j) * GSTR + em]);
                    const float go = sigm  (gatesT[(192 + j) * GSTR + em]);
                    const float c  = gf * creg[q] + gi * gg;
                    creg[q] = c;
                    const float h = go * tanh_f(c);
                    hbuf[em][j] = f2bf(h);             // next step's A operand
                    if (t == T_SEQ - 1) hF[em][j] = h; // only needed for final projection
                }
            }
            // commit x_{t+1} from the staged pipeline (pure DS read, no global wait)
            if (isx && t + 1 < T_SEQ)
                hbuf[xm][HID + xi] = f2bf(xstage[par][e8][tid]);
        }

        // retire the in-flight chunk into the other parity (loadcnt wait lands here,
        // ~XCH steps after the loads were issued; thread-private columns, no barrier)
        if (isx) {
#pragma unroll
            for (int e = 0; e < XCH; ++e) xstage[par ^ 1][e][tid] = xl[e];
        }
    }

    // ---------------- final projection: out = h_T @ fc_w^T + fc_b ----------------
    __syncthreads();
    if (tid < MB) {
        float acc = fc_b[0];
        for (int j = 0; j < HID; ++j) acc += hF[tid][j] * fc_w[j];
        out[bBase + tid] = acc;
    }
}

extern "C" void kernel_launch(void* const* d_in, const int* in_sizes, int n_in,
                              void* d_out, int out_size, void* d_ws, size_t ws_size,
                              hipStream_t stream)
{
    (void)in_sizes; (void)n_in; (void)out_size; (void)d_ws; (void)ws_size;
    const float* x   = (const float*)d_in[0];
    const float* Wih = (const float*)d_in[1];
    const float* Whh = (const float*)d_in[2];
    const float* bih = (const float*)d_in[3];
    const float* bhh = (const float*)d_in[4];
    const float* fcw = (const float*)d_in[5];
    const float* fcb = (const float*)d_in[6];
    lstm_persist<<<B_SZ / MB, 256, 0, stream>>>(x, Wih, Whh, bih, bhh, fcw, fcb, (float*)d_out);
}
